// Enc_Dec_linear_38250978738658
// MI455X (gfx1250) — matrix operand order repaired and checked by reading, not compiled, not run on hardware
//
#include <hip/hip_runtime.h>
#include <hip/hip_bf16.h>

// Problem constants (from reference)
#define Bb   16
#define Pp   12
#define Nn   10000
#define Ff   64
#define Tt   12
#define DOo  64
#define ROWS (Bb * Nn)          // 160000 independent rows
#define WAVES_PER_BLOCK 8       // 256 threads, wave32
#define NBLOCKS (ROWS / 16 / WAVES_PER_BLOCK)  // 1250

typedef __attribute__((ext_vector_type(16))) __bf16 v16bf;
typedef __attribute__((ext_vector_type(8)))  float  v8f;

// Fast transcendental helpers: v_exp_f32 + raw v_rcp_f32 (no IEEE div sequence).
__device__ __forceinline__ float fast_rcp(float x){ return __builtin_amdgcn_rcpf(x); }
__device__ __forceinline__ float sigf(float x)  { return fast_rcp(1.0f + __expf(-x)); }
__device__ __forceinline__ float tanhf_(float x){ return 2.0f * fast_rcp(1.0f + __expf(-2.0f * x)) - 1.0f; }
__device__ __forceinline__ __bf16 bfc(float x)  { return (__bf16)x; }

// ---------------------------------------------------------------------------
// LDS: all weights bf16 *transposed* (WT[n*ldk + k]) so B-fragments read
// contiguous K pairs; per-wave 16x128 bf16 activation staging buffer.
// Total ~147 KB; two blocks (16 waves) fit per 320 KB WGP.
// ---------------------------------------------------------------------------
struct SharedMem {
    __bf16 WTeg[128 * 128];   // eg_w  (2F x 2F) transposed, ldk=128
    __bf16 WTeu[ 64 * 128];   // eu_w  (2F x F)  transposed, ldk=128
    __bf16 WTeW[ 64 *  64];   // eW    (F x F)   transposed, ldk=64
    __bf16 WTdg[128 * 128];   // dg_w
    __bf16 WTdu[ 64 * 128];   // du_w
    __bf16 WTdW[ 64 *  64];   // dW
    float  egb[128], eub[64], ebv[64];
    float  dgb[128], dub[64], dbv[64];
    float  winw[64], winb[64], woutw[64], woutb;
    __bf16 Abuf[WAVES_PER_BLOCK][16 * 128];  // per-wave [x | h] staging
};

// A fragment (16x32 bf16): M = lane%16; VGPR v holds K = 16*(v/4) + 8*(lane/16) + 2*(v%4) (+1)
__device__ __forceinline__ v16bf load_A(const __bf16* Ab, int kt, int lane) {
    v16bf a;
    const __bf16* row = Ab + (lane & 15) * 128 + kt * 32 + ((lane >> 4) << 3);
#pragma unroll
    for (int v = 0; v < 8; ++v) {
        int k = ((v >> 2) << 4) + ((v & 3) << 1);
        a[2 * v]     = row[k];
        a[2 * v + 1] = row[k + 1];
    }
    return a;
}

// B fragment (32x16 bf16) from transposed weight WT[n*ldk + k]:
// N = lane%16; VGPR v holds K = 16*(v/4) + 8*(lane/16) + 2*(v%4) (+1)
__device__ __forceinline__ v16bf load_B(const __bf16* WT, int ldk, int kt, int nt, int lane) {
    v16bf b;
    const __bf16* col = WT + (nt * 16 + (lane & 15)) * ldk + kt * 32 + ((lane >> 4) << 3);
#pragma unroll
    for (int v = 0; v < 8; ++v) {
        int k = ((v >> 2) << 4) + ((v & 3) << 1);
        b[2 * v]     = col[k];
        b[2 * v + 1] = col[k + 1];
    }
    return b;
}

// Store C-layout tile (v8f: reg r -> row r + 8*(lane/16), col lane%16) into Abuf column-tile ct
__device__ __forceinline__ void store_tile(__bf16* Ab, int ct, v8f c, int lane) {
    int col   = ct * 16 + (lane & 15);
    int rbase = (lane >> 4) << 3;
#pragma unroll
    for (int r = 0; r < 8; ++r) Ab[(rbase + r) * 128 + col] = bfc(c[r]);
}

__device__ __forceinline__ void stageT(const float* W, __bf16* WT, int K, int Ncol, int tid, int nth) {
    for (int i = tid; i < K * Ncol; i += nth) {
        int k = i / Ncol, n = i - k * Ncol;
        WT[n * K + k] = bfc(W[i]);
    }
}
__device__ __forceinline__ void stageV(const float* s, float* d, int n, int tid, int nth) {
    for (int i = tid; i < n; i += nth) d[i] = s[i];
}

// One GRU step, streamed to minimize live accumulators.
// Abuf holds [x(0:64) | h(64:128)]. hx = state in C layout (4 N-tiles).
__device__ __forceinline__ void gru_step(const __bf16* WTg, const float* gb,
                                         const __bf16* WTu, const float* ub,
                                         __bf16* Ab, v8f hx[4], int lane) {
    v16bf af[4];
#pragma unroll
    for (int kt = 0; kt < 4; ++kt) af[kt] = load_A(Ab, kt, lane);

    // ---- r tiles (gate cols 0:64): consume immediately -> Abuf h-region := r*h ----
    // (safe: A fragments for the remaining matmuls are already in registers)
#pragma unroll
    for (int nt = 0; nt < 4; ++nt) {
        v8f acc = {0.f, 0.f, 0.f, 0.f, 0.f, 0.f, 0.f, 0.f};
#pragma unroll
        for (int kt = 0; kt < 4; ++kt) {
            v16bf bf = load_B(WTg, 128, kt, nt, lane);
            acc = __builtin_amdgcn_wmma_f32_16x16x32_bf16(false, af[kt], false, bf,
                                                          (short)0, acc, false, false);
        }
        float bias = gb[nt * 16 + (lane & 15)];
        v8f rh;
#pragma unroll
        for (int i = 0; i < 8; ++i) rh[i] = sigf(acc[i] + bias) * hx[nt][i];
        store_tile(Ab, 4 + nt, rh, lane);
    }

    // ---- u tiles (gate cols 64:128): keep only u ----
    v8f u[4];
#pragma unroll
    for (int nt = 0; nt < 4; ++nt) {
        v8f acc = {0.f, 0.f, 0.f, 0.f, 0.f, 0.f, 0.f, 0.f};
#pragma unroll
        for (int kt = 0; kt < 4; ++kt) {
            v16bf bf = load_B(WTg, 128, kt, nt + 4, lane);
            acc = __builtin_amdgcn_wmma_f32_16x16x32_bf16(false, af[kt], false, bf,
                                                          (short)0, acc, false, false);
        }
        float bias = gb[(nt + 4) * 16 + (lane & 15)];
#pragma unroll
        for (int i = 0; i < 8; ++i) u[nt][i] = sigf(acc[i] + bias);
    }

    // reload A for h-region (now holds r*h; x half unchanged)
    af[2] = load_A(Ab, 2, lane);
    af[3] = load_A(Ab, 3, lane);

    // ---- cy tiles: consume immediately into h' = u*h + (1-u)*cy ----
#pragma unroll
    for (int nt = 0; nt < 4; ++nt) {
        v8f acc = {0.f, 0.f, 0.f, 0.f, 0.f, 0.f, 0.f, 0.f};
#pragma unroll
        for (int kt = 0; kt < 4; ++kt) {
            v16bf bf = load_B(WTu, 128, kt, nt, lane);
            acc = __builtin_amdgcn_wmma_f32_16x16x32_bf16(false, af[kt], false, bf,
                                                          (short)0, acc, false, false);
        }
        float bias = ub[nt * 16 + (lane & 15)];
#pragma unroll
        for (int i = 0; i < 8; ++i) {
            float cy = tanhf_(acc[i] + bias);
            hx[nt][i] = u[nt][i] * hx[nt][i] + (1.0f - u[nt][i]) * cy;
        }
        store_tile(Ab, 4 + nt, hx[nt], lane);
    }
}

__global__ __launch_bounds__(WAVES_PER_BLOCK * 32)
__attribute__((amdgpu_num_vgpr(256)))
void enc_dec_gru_fused(const float* __restrict__ inputs,
                       const float* __restrict__ Win_w, const float* __restrict__ Win_b,
                       const float* __restrict__ eg_w,  const float* __restrict__ eg_b,
                       const float* __restrict__ eu_w,  const float* __restrict__ eu_b,
                       const float* __restrict__ eW,    const float* __restrict__ eb,
                       const float* __restrict__ dg_w,  const float* __restrict__ dg_b,
                       const float* __restrict__ du_w,  const float* __restrict__ du_b,
                       const float* __restrict__ dW,    const float* __restrict__ db,
                       const float* __restrict__ Wout_w,const float* __restrict__ Wout_b,
                       float* __restrict__ out) {
    __shared__ SharedMem sm;
    const int tid = threadIdx.x, nth = blockDim.x;

    // ---- stage all weights (bf16, transposed) + biases into LDS ----
    stageT(eg_w, sm.WTeg, 128, 128, tid, nth);
    stageT(eu_w, sm.WTeu, 128,  64, tid, nth);
    stageT(eW,   sm.WTeW,  64,  64, tid, nth);
    stageT(dg_w, sm.WTdg, 128, 128, tid, nth);
    stageT(du_w, sm.WTdu, 128,  64, tid, nth);
    stageT(dW,   sm.WTdW,  64,  64, tid, nth);
    stageV(eg_b, sm.egb, 128, tid, nth);
    stageV(eu_b, sm.eub,  64, tid, nth);
    stageV(eb,   sm.ebv,  64, tid, nth);
    stageV(dg_b, sm.dgb, 128, tid, nth);
    stageV(du_b, sm.dub,  64, tid, nth);
    stageV(db,   sm.dbv,  64, tid, nth);
    stageV(Win_w,  sm.winw, 64, tid, nth);
    stageV(Win_b,  sm.winb, 64, tid, nth);
    stageV(Wout_w, sm.woutw,64, tid, nth);
    if (tid == 0) sm.woutb = Wout_b[0];
    __syncthreads();

    const int wave = tid >> 5, lane = tid & 31;
    const int row0 = (blockIdx.x * WAVES_PER_BLOCK + wave) * 16;
    __bf16* Ab = sm.Abuf[wave];

    // zero the h-region of the staging buffer
    for (int i = lane; i < 16 * 64; i += 32) {
        int r = i >> 6, c = i & 63;
        Ab[r * 128 + 64 + c] = bfc(0.0f);
    }

    v8f hx[4];
#pragma unroll
    for (int nt = 0; nt < 4; ++nt) hx[nt] = (v8f){0.f, 0.f, 0.f, 0.f, 0.f, 0.f, 0.f, 0.f};

    // per-lane row this half-wave services for x staging (lanes l and l+16 share a row)
    const int grow = row0 + (lane & 15);
    const int gbat = grow / Nn;
    const int gn   = grow - gbat * Nn;

    // ---------------- Encoder: scan over P steps ----------------
#pragma unroll 1
    for (int p = 0; p < Pp; ++p) {
        float s = inputs[(gbat * Pp + p) * Nn + gn];  // x[b,p,n,0]
        const int r = lane & 15, cb = (lane >> 4) * 32;
#pragma unroll
        for (int c = 0; c < 32; ++c) {
            int f = cb + c;
            Ab[r * 128 + f] = bfc(s * sm.winw[f] + sm.winb[f]);
        }
        gru_step(sm.WTeg, sm.egb, sm.WTeu, sm.eub, Ab, hx, lane);
    }

    // ---------------- yt_enc = sigmoid(h_enc @ eW + eb) -> Abuf x-region ----------------
    {
        v16bf af[2];
        af[0] = load_A(Ab, 2, lane);
        af[1] = load_A(Ab, 3, lane);
#pragma unroll
        for (int nt = 0; nt < 4; ++nt) {
            v8f acc = {0.f, 0.f, 0.f, 0.f, 0.f, 0.f, 0.f, 0.f};
#pragma unroll
            for (int kt = 0; kt < 2; ++kt) {
                v16bf bf = load_B(sm.WTeW, 64, kt, nt, lane);
                acc = __builtin_amdgcn_wmma_f32_16x16x32_bf16(false, af[kt], false, bf,
                                                              (short)0, acc, false, false);
            }
            float bias = sm.ebv[nt * 16 + (lane & 15)];
            v8f yt;
#pragma unroll
            for (int i = 0; i < 8; ++i) yt[i] = sigf(acc[i] + bias);
            store_tile(Ab, nt, yt, lane);   // x-region now holds yt_enc for all decoder steps
        }
    }

    // ---------------- Decoder: T steps, fused output head ----------------
#pragma unroll 1
    for (int t = 0; t < Tt; ++t) {
        gru_step(sm.WTdg, sm.dgb, sm.WTdu, sm.dub, Ab, hx, lane);

        // yt = sigmoid(hy @ dW + db); out = yt @ Wout_w + Wout_b (dot fused)
        v16bf af[2];
        af[0] = load_A(Ab, 2, lane);
        af[1] = load_A(Ab, 3, lane);
        v8f oacc = {0.f, 0.f, 0.f, 0.f, 0.f, 0.f, 0.f, 0.f};
#pragma unroll
        for (int nt = 0; nt < 4; ++nt) {
            v8f acc = {0.f, 0.f, 0.f, 0.f, 0.f, 0.f, 0.f, 0.f};
#pragma unroll
            for (int kt = 0; kt < 2; ++kt) {
                v16bf bf = load_B(sm.WTdW, 64, kt, nt, lane);
                acc = __builtin_amdgcn_wmma_f32_16x16x32_bf16(false, af[kt], false, bf,
                                                              (short)0, acc, false, false);
            }
            float bias = sm.dbv[nt * 16 + (lane & 15)];
            float wo   = sm.woutw[nt * 16 + (lane & 15)];
#pragma unroll
            for (int i = 0; i < 8; ++i) oacc[i] += sigf(acc[i] + bias) * wo;
        }
        // reduce the 16 N-lanes of each half-wave; lanes 0/16 own rows i and 8+i
#pragma unroll
        for (int i = 0; i < 8; ++i) {
            float v = oacc[i];
            v += __shfl_xor(v, 1);
            v += __shfl_xor(v, 2);
            v += __shfl_xor(v, 4);
            v += __shfl_xor(v, 8);
            if ((lane & 15) == 0) {
                int row = i + ((lane >> 4) << 3);
                int g   = row0 + row;
                int bb  = g / Nn;
                int nn2 = g - bb * Nn;
                out[(bb * Tt + t) * Nn + nn2] = v + sm.woutb;
            }
        }
    }
}

extern "C" void kernel_launch(void* const* d_in, const int* in_sizes, int n_in,
                              void* d_out, int out_size, void* d_ws, size_t ws_size,
                              hipStream_t stream) {
    (void)in_sizes; (void)n_in; (void)out_size; (void)d_ws; (void)ws_size;
    const float* inputs = (const float*)d_in[0];
    const float* Win_w  = (const float*)d_in[1];
    const float* Win_b  = (const float*)d_in[2];
    const float* eg_w   = (const float*)d_in[3];
    const float* eg_b   = (const float*)d_in[4];
    const float* eu_w   = (const float*)d_in[5];
    const float* eu_b   = (const float*)d_in[6];
    const float* eW     = (const float*)d_in[7];
    const float* eb     = (const float*)d_in[8];
    const float* dg_w   = (const float*)d_in[9];
    const float* dg_b   = (const float*)d_in[10];
    const float* du_w   = (const float*)d_in[11];
    const float* du_b   = (const float*)d_in[12];
    const float* dW     = (const float*)d_in[13];
    const float* db     = (const float*)d_in[14];
    const float* Wout_w = (const float*)d_in[15];
    const float* Wout_b = (const float*)d_in[16];
    float* out = (float*)d_out;

    enc_dec_gru_fused<<<NBLOCKS, WAVES_PER_BLOCK * 32, 0, stream>>>(
        inputs, Win_w, Win_b, eg_w, eg_b, eu_w, eu_b, eW, eb,
        dg_w, dg_b, du_w, du_b, dW, db, Wout_w, Wout_b, out);
}
